// VisionSdpaAttention_49246095016313
// MI455X (gfx1250) — matrix operand order repaired and checked by reading, not compiled, not run on hardware
//
#include <hip/hip_runtime.h>
#include <hip/hip_bf16.h>

// ---------------------------------------------------------------------------
// Varlen vision SDPA (flash-attention) for gfx1250 / MI455X.
//   q,k,v: [3072, 16, 80] f32 ; out: [3072, 16, 80] f32
//   block-diagonal mask from cu_seqlens[5]
// Block = 4 waves, one head, 64 query rows (16 per wave). 32-key tiles are
// staged cooperatively into LDS as f16 (K row-major, V transposed) with
// double buffering (one barrier/tile); WMMA fragments are then pure
// ds_load_b128. f16 WMMA 16x16x32 for QK^T and P*V; f32 online softmax in
// exp2 domain; ds_swizzle xor reductions.
// ---------------------------------------------------------------------------

typedef __attribute__((ext_vector_type(16))) _Float16 v16h;
typedef __attribute__((ext_vector_type(8)))  _Float16 v8h;
typedef __attribute__((ext_vector_type(4)))  _Float16 v4h;
typedef __attribute__((ext_vector_type(2)))  _Float16 v2h;
typedef __attribute__((ext_vector_type(8)))  float    v8f;
typedef __attribute__((ext_vector_type(4)))  float    f32x4;

#define S_TOT   3072
#define NH      16
#define HD      80
#define TSTRIDE (NH * HD)      // 1280 floats per token
#define WAVES   4              // waves (q-tiles) per workgroup
#define NTHREADS (WAVES * 32)
#define NEG_INF (-3.0e38f)
#define LOG2E   1.44269504088896340736f

__device__ __forceinline__ int seq_of(int pos, int c1, int c2, int c3) {
  return (pos >= c1) + (pos >= c2) + (pos >= c3);
}

// xor-lane exchange within 16-lane halves via ds_swizzle (immediate pattern).
template <int MASK>
__device__ __forceinline__ float swz_xor(float x) {
  return __int_as_float(
      __builtin_amdgcn_ds_swizzle(__float_as_int(x), (MASK << 10) | 0x1f));
}
__device__ __forceinline__ float rowmax16(float x) {
  x = fmaxf(x, swz_xor<1>(x));
  x = fmaxf(x, swz_xor<2>(x));
  x = fmaxf(x, swz_xor<4>(x));
  x = fmaxf(x, swz_xor<8>(x));
  return x;
}
__device__ __forceinline__ float rowsum16(float x) {
  x += swz_xor<1>(x);
  x += swz_xor<2>(x);
  x += swz_xor<4>(x);
  x += swz_xor<8>(x);
  return x;
}

// Cooperative staging of one 32-key tile (128 threads):
//   kbuf : K  [token 0..31][dim 0..79]  f16 (row-major)
//   vtbuf: V^T [dim 0..79][token 0..31] f16 (transposed)
__device__ __forceinline__ void stage_tile(const float* __restrict__ kp,
                                           const float* __restrict__ vp,
                                           int t0, int h, int tid,
                                           _Float16* kbuf, _Float16* vtbuf) {
  {
    // K: 4 threads per token, 20 dims each (coalesced float4 loads).
    const int tok = tid >> 2;
    const int d0  = (tid & 3) * 20;
    const float* src = kp + (size_t)(t0 + tok) * TSTRIDE + h * HD + d0;
    _Float16* dst = kbuf + tok * HD + d0;
#pragma unroll
    for (int j = 0; j < 20; j += 4) {
      f32x4 f = *reinterpret_cast<const f32x4*>(src + j);
      v4h hv;
      hv[0] = (_Float16)f.x; hv[1] = (_Float16)f.y;
      hv[2] = (_Float16)f.z; hv[3] = (_Float16)f.w;
      *reinterpret_cast<v4h*>(dst + j) = hv;   // 8B aligned
    }
  }
  {
    // V^T: thread owns token pair (2p,2p+1) x 10 dims; packs pairs -> b32.
    const int p  = tid >> 3;          // 0..15
    const int d0 = (tid & 7) * 10;    // 0..70
    const float* s0 = vp + (size_t)(t0 + 2 * p)     * TSTRIDE + h * HD + d0;
    const float* s1 = vp + (size_t)(t0 + 2 * p + 1) * TSTRIDE + h * HD + d0;
#pragma unroll
    for (int j = 0; j < 10; ++j) {
      v2h hv;
      hv[0] = (_Float16)s0[j];
      hv[1] = (_Float16)s1[j];
      *reinterpret_cast<v2h*>(vtbuf + (d0 + j) * 32 + 2 * p) = hv;
    }
  }
}

// B-fragment for QK^T from staged K: keys nt*16.., K-dim chunk c.
// Lane l: column n=l&15 (token), dims c*32+(l>>4)*8+{0..7} and c*32+16+(l>>4)*8+{0..7} -> 2x b128.
__device__ __forceinline__ v16h kfrag_lds(const _Float16* kbuf, int nt, int c,
                                          int lane) {
  v16h b;
  const int n = lane & 15, half = lane >> 4;
  const int d0 = c * 32 + half * 8;
  const _Float16* src = kbuf + (nt * 16 + n) * HD + d0;   // 16B aligned
  if (d0 < HD) {
    v8h lo = *reinterpret_cast<const v8h*>(src);
#pragma unroll
    for (int j = 0; j < 8; ++j) b[j] = lo[j];
  } else {
#pragma unroll
    for (int j = 0; j < 8; ++j) b[j] = (_Float16)0.0f;
  }
  if (d0 + 16 < HD) {
    v8h hi = *reinterpret_cast<const v8h*>(src + 16);
#pragma unroll
    for (int j = 0; j < 8; ++j) b[8 + j] = hi[j];
  } else {
#pragma unroll
    for (int j = 0; j < 8; ++j) b[8 + j] = (_Float16)0.0f;
  }
  return b;
}

// B-fragment for P*V from staged V^T: out-dim column n=l&15 of tile t,
// tokens (l>>4)*16+{0..15} contiguous in V^T -> 2x b128.
__device__ __forceinline__ v16h vfrag_lds(const _Float16* vtbuf, int t,
                                          int lane) {
  const int n = lane & 15, half = lane >> 4;
  const _Float16* src = vtbuf + (t * 16 + n) * 32 + half * 16;  // 32B aligned
  v8h lo = *reinterpret_cast<const v8h*>(src);
  v8h hi = *reinterpret_cast<const v8h*>(src + 8);
  v16h b;
#pragma unroll
  for (int j = 0; j < 8; ++j) { b[j] = lo[j]; b[8 + j] = hi[j]; }
  return b;
}

// A-fragment for QK^T: Q rows q0..q0+15, K-dims chunk c (global, read once).
__device__ __forceinline__ v16h load_qfrag(const float* __restrict__ qp,
                                           int q0, int h, int c, int lane,
                                           float scale) {
  v16h a;
  const int m = lane & 15, half = lane >> 4;
  const float* base = qp + (size_t)(q0 + m) * TSTRIDE + h * HD;
  const int dlo = c * 32 + half * 8;        // always < 80 for c<=2
  const int dhi = c * 32 + 16 + half * 8;   // >= 80 for c==2 => zero pad
#pragma unroll
  for (int j = 0; j < 8; ++j) a[j] = (_Float16)(base[dlo + j] * scale);
  if (dhi < HD) {
#pragma unroll
    for (int j = 0; j < 8; ++j) a[8 + j] = (_Float16)(base[dhi + j] * scale);
  } else {
#pragma unroll
    for (int j = 0; j < 8; ++j) a[8 + j] = (_Float16)0.0f;
  }
  return a;
}

__global__ __launch_bounds__(NTHREADS)
void vis_sdpa_fa_kernel(const float* __restrict__ q,
                        const float* __restrict__ k,
                        const float* __restrict__ v,
                        const int*   __restrict__ cu,
                        float*       __restrict__ out) {
  __shared__ _Float16 kbuf[2][32 * HD];     // 2 x 5 KB
  __shared__ _Float16 vtbuf[2][HD * 32];    // 2 x 5 KB
  __shared__ _Float16 pbuf[WAVES * 512];    // per-wave 16x32 P (f16), 4 KB

  const int tid  = threadIdx.x;
  const int lane = tid & 31;
  const int wave = tid >> 5;
  const int half = lane >> 4;
  const int h    = blockIdx.y;
  const int q0   = (blockIdx.x * WAVES + wave) * 16;
  const int B0   = blockIdx.x * (WAVES * 16);   // block's first query row

  const int c0 = cu[0], c1 = cu[1], c2 = cu[2], c3 = cu[3], c4 = cu[4];

  // scale * log2(e): softmax runs in the exp2 domain (v_exp_f32 is base-2).
  const float qscale = rsqrtf((float)HD) * LOG2E;

  v16h qa0 = load_qfrag(q, q0, h, 0, lane, qscale);
  v16h qa1 = load_qfrag(q, q0, h, 1, lane, qscale);
  v16h qa2 = load_qfrag(q, q0, h, 2, lane, qscale);

  // Sequence id per query row held by this lane (C-layout rows r+8*half).
  int qsid[8];
#pragma unroll
  for (int r = 0; r < 8; ++r) qsid[r] = seq_of(q0 + r + 8 * half, c1, c2, c3);

  // Per-wave key range (wave-uniform).
  const int sa = seq_of(q0,      c1, c2, c3);
  const int sb = seq_of(q0 + 15, c1, c2, c3);
  const int kw0 = ((sa == 0) ? c0 : (sa == 1) ? c1 : (sa == 2) ? c2 : c3) & ~31;
  const int kw1 = (sb == 0) ? c1 : (sb == 1) ? c2 : (sb == 2) ? c3 : c4;

  // Block-uniform union key range over the 4 waves.
  const int sa_b = seq_of(B0,      c1, c2, c3);
  const int sb_b = seq_of(B0 + WAVES * 16 - 1, c1, c2, c3);
  const int ku0 = ((sa_b == 0) ? c0 : (sa_b == 1) ? c1 : (sa_b == 2) ? c2 : c3) & ~31;
  const int ku1 = (sb_b == 0) ? c1 : (sb_b == 1) ? c2 : (sb_b == 2) ? c3 : c4;

  float mrun[8], lrun[8];
#pragma unroll
  for (int r = 0; r < 8; ++r) { mrun[r] = NEG_INF; lrun[r] = 0.0f; }
  v8f zero8 = {};
  v8f oacc[5];
#pragma unroll
  for (int t = 0; t < 5; ++t) oacc[t] = zero8;

  _Float16* pb = &pbuf[wave * 512];

  // Prologue: stage first tile into buffer 0.
  stage_tile(k, v, ku0, h, tid, kbuf[0], vtbuf[0]);

  int buf = 0;
  for (int t0 = ku0; t0 < ku1; t0 += 32, buf ^= 1) {
    __syncthreads();                       // staged buf ready; prev reads done

    if (t0 + 64 < ku1) {                   // spread prefetch of tile-after-next
      __builtin_prefetch(k + (size_t)(t0 + 64 + (tid >> 2)) * TSTRIDE
                           + h * HD + (tid & 3) * 16, 0, 1);
      __builtin_prefetch(v + (size_t)(t0 + 64 + (tid >> 2)) * TSTRIDE
                           + h * HD + (tid & 3) * 16, 0, 1);
    }
    if (t0 + 32 < ku1)                     // stage next tile into other buffer
      stage_tile(k, v, t0 + 32, h, tid, kbuf[buf ^ 1], vtbuf[buf ^ 1]);

    if (t0 >= kw0 && t0 < kw1) {           // wave-uniform: tile in my range
      const _Float16* kb_ = kbuf[buf];
      const _Float16* vt_ = vtbuf[buf];

      // ---- S = (Q*scale) . K^T : two 16x16 tiles ----
      v8f s0 = {}, s1 = {};
      s0 = __builtin_amdgcn_wmma_f32_16x16x32_f16(false, qa0, false,
              kfrag_lds(kb_, 0, 0, lane), (short)0, s0, false, false);
      s0 = __builtin_amdgcn_wmma_f32_16x16x32_f16(false, qa1, false,
              kfrag_lds(kb_, 0, 1, lane), (short)0, s0, false, false);
      s0 = __builtin_amdgcn_wmma_f32_16x16x32_f16(false, qa2, false,
              kfrag_lds(kb_, 0, 2, lane), (short)0, s0, false, false);
      s1 = __builtin_amdgcn_wmma_f32_16x16x32_f16(false, qa0, false,
              kfrag_lds(kb_, 1, 0, lane), (short)0, s1, false, false);
      s1 = __builtin_amdgcn_wmma_f32_16x16x32_f16(false, qa1, false,
              kfrag_lds(kb_, 1, 1, lane), (short)0, s1, false, false);
      s1 = __builtin_amdgcn_wmma_f32_16x16x32_f16(false, qa2, false,
              kfrag_lds(kb_, 1, 2, lane), (short)0, s1, false, false);

      // ---- mask + online softmax (f32, exp2 domain) ----
      const int ksid0 = seq_of(t0 +      (lane & 15), c1, c2, c3);
      const int ksid1 = seq_of(t0 + 16 + (lane & 15), c1, c2, c3);
      float alpha[8];
#pragma unroll
      for (int r = 0; r < 8; ++r) {
        const bool m0 = (qsid[r] != ksid0);
        const bool m1 = (qsid[r] != ksid1);
        const float sv0 = m0 ? NEG_INF : s0[r];
        const float sv1 = m1 ? NEG_INF : s1[r];
        const float tm = rowmax16(fmaxf(sv0, sv1));   // row max over 32 keys
        const float mn = fmaxf(mrun[r], tm);
        const float al = (mn <= -1.0e37f) ? 1.0f : exp2f(mrun[r] - mn);
        const float pv0 = m0 ? 0.0f : exp2f(sv0 - mn);
        const float pv1 = m1 ? 0.0f : exp2f(sv1 - mn);
        const float rs = rowsum16(pv0 + pv1);         // row sum over 32 keys
        lrun[r] = lrun[r] * al + rs;
        mrun[r] = mn;
        alpha[r] = al;
        // C-layout P (f16) -> per-wave bounce buffer
        pb[(r + 8 * half) * 32 +      (lane & 15)] = (_Float16)pv0;
        pb[(r + 8 * half) * 32 + 16 + (lane & 15)] = (_Float16)pv1;
      }
      __asm__ volatile("s_wait_dscnt 0x0" ::: "memory");

      // A-layout P: row l&15, keys half*16..+15 -> 2x b128.
      v16h pa;
      {
        const _Float16* row = pb + (lane & 15) * 32 + half * 16;
        v8h lo = *reinterpret_cast<const v8h*>(row);
        v8h hi = *reinterpret_cast<const v8h*>(row + 8);
#pragma unroll
        for (int j = 0; j < 8; ++j) { pa[j] = lo[j]; pa[8 + j] = hi[j]; }
      }

      // ---- O = O*alpha + P.V over 5 output-dim tiles (K=32 full) ----
#pragma unroll
      for (int t = 0; t < 5; ++t) {
        v16h vb = vfrag_lds(vt_, t, lane);
        v8f o = oacc[t];
#pragma unroll
        for (int r = 0; r < 8; ++r) o[r] = o[r] * alpha[r];
        o = __builtin_amdgcn_wmma_f32_16x16x32_f16(false, pa, false, vb,
                                                   (short)0, o, false, false);
        oacc[t] = o;
      }
    }
  }

  // ---- finalize: O / l, write out ----
  float inv[8];
#pragma unroll
  for (int r = 0; r < 8; ++r) inv[r] = (lrun[r] > 0.0f) ? 1.0f / lrun[r] : 0.0f;
#pragma unroll
  for (int t = 0; t < 5; ++t) {
#pragma unroll
    for (int r = 0; r < 8; ++r) {
      out[(size_t)(q0 + r + 8 * half) * TSTRIDE + h * HD + t * 16 + (lane & 15)]
          = oacc[t][r] * inv[r];
    }
  }
}

extern "C" void kernel_launch(void* const* d_in, const int* in_sizes, int n_in,
                              void* d_out, int out_size, void* d_ws, size_t ws_size,
                              hipStream_t stream) {
  const float* q  = (const float*)d_in[0];
  const float* k  = (const float*)d_in[1];
  const float* v  = (const float*)d_in[2];
  // d_in[3] = bsz (==1), unused
  const int*   cu = (const int*)d_in[4];
  float* out = (float*)d_out;

  dim3 grid(S_TOT / (16 * WAVES), NH);   // 48 x 16
  dim3 block(NTHREADS);                  // 128 threads = 4 waves
  hipLaunchKernelGGL(vis_sdpa_fa_kernel, grid, block, 0, stream,
                     q, k, v, cu, out);
}
